// BlockConvolutionLean_74457553044363
// MI455X (gfx1250) — matrix operand order repaired and checked by reading, not compiled, not run on hardware
//
#include <hip/hip_runtime.h>

// Types for WMMA operands / descriptors
typedef __attribute__((ext_vector_type(16))) _Float16 v16h;
typedef __attribute__((ext_vector_type(8)))  _Float16 v8h;
typedef __attribute__((ext_vector_type(8)))  float    v8f;
typedef __attribute__((ext_vector_type(4)))  float    v4f;
typedef __attribute__((ext_vector_type(4)))  unsigned int v4u;
typedef __attribute__((ext_vector_type(8)))  int      v8i;
typedef __attribute__((ext_vector_type(4)))  int      v4i;

#define TDIM 256       // output feature dim (rows of W)
#define EDIM 256       // input feature dim (K)
#define BSZ  8         // block size for causal cumsum
#define MTOT 65536     // N*S rows
#define ROWS_PER_WG 128
#define LDS_STRIDE 80   // f16 B-buffer row: 32 halfs (64B) + 16B pad (20-dword stride)
#define STAGE_STRIDE 144 // f32 TDM staging row: 32 floats (128B) + 16B pad (36-dword stride)

#if defined(__gfx1250__) && __has_builtin(__builtin_amdgcn_tensor_load_to_lds)
#define USE_TDM 1
#else
#define USE_TDM 0
#endif

__global__ __launch_bounds__(256)
void block_conv_lean_kernel(const float* __restrict__ X,
                            const float* __restrict__ W,
                            const float* __restrict__ bias,
                            float* __restrict__ out) {
  // f16 B operand buffer: 256 rows x 80B = 20 KB
  __shared__ __align__(16) unsigned char ldsW[256 * LDS_STRIDE];
#if USE_TDM
  // raw f32 W chunk staged by the Tensor Data Mover: 256 rows x 144B = 36 KB
  __shared__ __align__(16) unsigned char ldsStage[256 * STAGE_STRIDE];
#endif

  const int tid     = threadIdx.x;
  const int wave    = tid >> 5;        // 8 wave32 waves per WG
  const int lane    = tid & 31;
  const int laneLo  = lane & 15;
  const int hi      = lane >> 4;
  const int rowBase = blockIdx.x * ROWS_PER_WG + wave * 16;

  // Effective bias: position r gets bias[r]; position 0 gets bias[0] twice.
  float beff[BSZ];
  #pragma unroll
  for (int r = 0; r < BSZ; ++r) beff[r] = bias[r];
  beff[0] += bias[0];

  v8f acc[16];
  #pragma unroll
  for (int t = 0; t < 16; ++t) { v8f z = {}; acc[t] = z; }

  const int m = rowBase + laneLo;      // A-matrix row owned by this lane

  for (int k0 = 0; k0 < EDIM; k0 += 32) {
    __syncthreads();  // previous iteration's LDS consumers are done

#if USE_TDM
    // --- Tensor Data Mover: DMA W[:, k0:k0+32) (f32) into ldsStage ---
    if (wave == 0) {
      const unsigned long long gaddr =
          (unsigned long long)(uintptr_t)W + (unsigned long long)k0 * 4ull;
      const unsigned int ldsOff = (unsigned int)(uintptr_t)(void*)ldsStage;
      v4u g0;
      g0[0] = 1u;                                    // count=1 (valid descriptor)
      g0[1] = ldsOff;                                // lds_addr
      g0[2] = (unsigned int)(gaddr & 0xFFFFFFFFull); // global_addr[31:0]
      g0[3] = (unsigned int)((gaddr >> 32) & 0x1FFFFFFull) | (2u << 30); // +type=2
      v8i g1;
      // data_size=4B(2), pad_enable, pad_interval=32 dwords(4), pad_amount=4 dwords(3)
      g1[0] = (int)((2u << 16) | (1u << 20) | (4u << 22) | (3u << 25));
      g1[1] = (int)(32u  << 16);   // tensor_dim0 = 32
      g1[2] = (int)(256u << 16);   // tensor_dim1 = 256
      g1[3] = (int)(32u  << 16);   // tile_dim0   = 32
      g1[4] = 256;                 // tile_dim1   = 256
      g1[5] = 256;                 // tensor_dim0_stride = 256
      g1[6] = 0;
      g1[7] = 0;
      v4i gz = {0, 0, 0, 0};
#if __clang_major__ >= 23
      v8i gz8 = {0, 0, 0, 0, 0, 0, 0, 0};
      __builtin_amdgcn_tensor_load_to_lds(g0, g1, gz, gz, gz8, 0);
#else
      __builtin_amdgcn_tensor_load_to_lds(g0, g1, gz, gz, 0);
#endif
      __builtin_amdgcn_s_wait_tensorcnt(0);
    }
    __syncthreads();  // staged f32 chunk visible to all waves

    // Convert staged f32 row -> padded f16 B row (one row per thread)
    {
      const v4f* sr = (const v4f*)(ldsStage + tid * STAGE_STRIDE);
      #pragma unroll
      for (int c = 0; c < 4; ++c) {
        v4f f0 = sr[2 * c + 0];
        v4f f1 = sr[2 * c + 1];
        v8h h;
        h[0] = (_Float16)f0[0]; h[1] = (_Float16)f0[1];
        h[2] = (_Float16)f0[2]; h[3] = (_Float16)f0[3];
        h[4] = (_Float16)f1[0]; h[5] = (_Float16)f1[1];
        h[6] = (_Float16)f1[2]; h[7] = (_Float16)f1[3];
        *(v8h*)(ldsW + tid * LDS_STRIDE + c * 16) = h;
      }
    }
#else
    // Fallback: stage W[:, k0:k0+32) from global, one row per thread
    {
      const v4f* wr = (const v4f*)(W + (size_t)tid * EDIM + k0);
      #pragma unroll
      for (int c = 0; c < 4; ++c) {
        v4f f0 = wr[2 * c + 0];
        v4f f1 = wr[2 * c + 1];
        v8h h;
        h[0] = (_Float16)f0[0]; h[1] = (_Float16)f0[1];
        h[2] = (_Float16)f0[2]; h[3] = (_Float16)f0[3];
        h[4] = (_Float16)f1[0]; h[5] = (_Float16)f1[1];
        h[6] = (_Float16)f1[2]; h[7] = (_Float16)f1[3];
        *(v8h*)(ldsW + tid * LDS_STRIDE + c * 16) = h;
      }
    }
#endif
    __syncthreads();

    // A fragment (16x32 f16): halfs 0..7 = K(k0+hi*8..+8), halfs 8..15 = K(+16)
    const float* xrow = X + (size_t)m * EDIM + k0 + hi * 8;
    v4f a0 = __builtin_nontemporal_load((const v4f*)xrow);
    v4f a1 = __builtin_nontemporal_load((const v4f*)xrow + 1);
    v4f a2 = __builtin_nontemporal_load((const v4f*)(xrow + 16));
    v4f a3 = __builtin_nontemporal_load((const v4f*)(xrow + 16) + 1);
    if (k0 + 32 < EDIM)
      __builtin_prefetch(xrow + 32, 0, 0);  // global_prefetch of next K chunk

    v16h a;
    a[0]  = (_Float16)a0[0]; a[1]  = (_Float16)a0[1];
    a[2]  = (_Float16)a0[2]; a[3]  = (_Float16)a0[3];
    a[4]  = (_Float16)a1[0]; a[5]  = (_Float16)a1[1];
    a[6]  = (_Float16)a1[2]; a[7]  = (_Float16)a1[3];
    a[8]  = (_Float16)a2[0]; a[9]  = (_Float16)a2[1];
    a[10] = (_Float16)a2[2]; a[11] = (_Float16)a2[3];
    a[12] = (_Float16)a3[0]; a[13] = (_Float16)a3[1];
    a[14] = (_Float16)a3[2]; a[15] = (_Float16)a3[3];

#define LOADB(tnv, dst) {                                                     \
    const v8h* bp = (const v8h*)(ldsW + ((tnv) * 16 + laneLo) * LDS_STRIDE    \
                                 + hi * 16);                                  \
    v8h blo = bp[0]; v8h bh2 = bp[2];                                         \
    dst = __builtin_shufflevector(blo, bh2,                                   \
          0, 1, 2, 3, 4, 5, 6, 7, 8, 9, 10, 11, 12, 13, 14, 15); }

    // 4 column tiles per step with distinct B fragments -> LDS/WMMA overlap
    #pragma unroll
    for (int tn = 0; tn < 16; tn += 4) {
      v16h b0, b1, b2, b3;
      LOADB(tn + 0, b0);
      LOADB(tn + 1, b1);
      LOADB(tn + 2, b2);
      LOADB(tn + 3, b3);
      acc[tn + 0] = __builtin_amdgcn_wmma_f32_16x16x32_f16(
          false, a, false, b0, (short)0, acc[tn + 0], false, false);
      acc[tn + 1] = __builtin_amdgcn_wmma_f32_16x16x32_f16(
          false, a, false, b1, (short)0, acc[tn + 1], false, false);
      acc[tn + 2] = __builtin_amdgcn_wmma_f32_16x16x32_f16(
          false, a, false, b2, (short)0, acc[tn + 2], false, false);
      acc[tn + 3] = __builtin_amdgcn_wmma_f32_16x16x32_f16(
          false, a, false, b3, (short)0, acc[tn + 3], false, false);
    }
#undef LOADB
  }

  // Epilogue: each lane's 8 acc registers are exactly one BS=8 block
  // (VGPR r -> row r for lanes 0-15, row 8+r for lanes 16-31): serial
  // exclusive prefix over registers + effective bias, NT stores.
  #pragma unroll
  for (int tn = 0; tn < 16; ++tn) {
    float* op = out + (size_t)(rowBase + hi * 8) * TDIM + tn * 16 + laneLo;
    float run = 0.0f;
    #pragma unroll
    for (int r = 0; r < BSZ; ++r) {
      __builtin_nontemporal_store(run + beff[r], op + (size_t)r * TDIM);
      run += acc[tn][r];
    }
  }
}

extern "C" void kernel_launch(void* const* d_in, const int* in_sizes, int n_in,
                              void* d_out, int out_size, void* d_ws, size_t ws_size,
                              hipStream_t stream) {
  (void)in_sizes; (void)n_in; (void)d_ws; (void)ws_size; (void)out_size;
  const float* X    = (const float*)d_in[0];  // seq_vector [8,8192,256] f32
  const float* W    = (const float*)d_in[1];  // [256,256] f32
  const float* bias = (const float*)d_in[2];  // [8] f32
  float* out = (float*)d_out;                 // [8,8192,256] f32

  dim3 grid(MTOT / ROWS_PER_WG);  // 512 workgroups
  dim3 block(256);                // 8 wave32 waves
  block_conv_lean_kernel<<<grid, block, 0, stream>>>(X, W, bias, out);
}
